// triGAT_41781441856240
// MI455X (gfx1250) — matrix operand order repaired and checked by reading, not compiled, not run on hardware
//
#include <hip/hip_runtime.h>

typedef __attribute__((ext_vector_type(16))) _Float16 v16h;
typedef __attribute__((ext_vector_type(8)))  _Float16 v8h;
typedef __attribute__((ext_vector_type(8)))  float    v8f;

#define NNODES 50000L
#define INF    64
#define HCF    128          // H*C = 8*16
#define HIDF   384          // 3*H*C

// ---------------- helpers ----------------
__device__ __forceinline__ unsigned fkey(float f) {
  unsigned u = __float_as_uint(f);
  return (u & 0x80000000u) ? ~u : (u | 0x80000000u);
}
__device__ __forceinline__ float fdec(unsigned k) {
  return (k & 0x80000000u) ? __uint_as_float(k & 0x7FFFFFFFu) : __uint_as_float(~k);
}
__device__ __forceinline__ float lrelu(float x) { return x > 0.f ? x : 0.2f * x; }

// ---------------- generic zero ----------------
__global__ void k_zero(unsigned* __restrict__ p, long n) {
  long i = (long)blockIdx.x * blockDim.x + threadIdx.x;
  if (i < n) p[i] = 0u;
}

// ---------------- f32 -> f16 conversions (one-time prep) ----------------
__global__ void k_cvt_x(const float* __restrict__ X, _Float16* __restrict__ Xh, long n) {
  long i = (long)blockIdx.x * blockDim.x + threadIdx.x;
  if (i < n) Xh[i] = (_Float16)X[i];
}

// W[64,128] row-major f32 -> Wt[128][64] col-major f16
__global__ void k_cvt_wt(const float* __restrict__ W, _Float16* __restrict__ Wt) {
  int id = blockIdx.x * blockDim.x + threadIdx.x;
  if (id >= INF * HCF) return;
  int k = id >> 7, n = id & 127;          // coalesced read over n
  Wt[n * INF + k] = (_Float16)W[id];
}

// ---------------- edge-feature mean reduction ----------------
__global__ void k_reduce_ea(const float* __restrict__ ea, long E, float* __restrict__ sums) {
  float s0 = 0.f, s1 = 0.f;
  for (long i = (long)blockIdx.x * blockDim.x + threadIdx.x; i < E;
       i += (long)gridDim.x * blockDim.x) {
    s0 += ea[2 * i];
    s1 += ea[2 * i + 1];
  }
  for (int o = 16; o > 0; o >>= 1) { s0 += __shfl_down(s0, o); s1 += __shfl_down(s1, o); }
  if ((threadIdx.x & 31) == 0) { atomicAdd(&sums[0], s0); atomicAdd(&sums[1], s1); }
}

// ---------------- fold We/a_e into per-head edge weights ----------------
// consts layout: [0..1]=ea sums, [2..17]=WeR11[d*8+h], [18..33]=WeR21, [34..35]=WeR12, [36..37]=WeR22
__global__ void k_wer(const float* We11, const float* ae11,
                      const float* We21, const float* ae21,
                      const float* We12, const float* ae12,
                      const float* We22, const float* ae22,
                      float* consts) {
  int t = threadIdx.x;
  if (t < 16) {
    int d = t >> 3, h = t & 7;
    float s1 = 0.f, s2 = 0.f;
    for (int c = 0; c < 16; ++c) {
      s1 += We11[d * HCF + h * 16 + c] * ae11[h * 16 + c];
      s2 += We21[d * HCF + h * 16 + c] * ae21[h * 16 + c];
    }
    consts[2 + t]  = s1;
    consts[18 + t] = s2;
  }
  if (t == 0) {
    consts[34] = We12[0] * ae12[0];
    consts[35] = We12[1] * ae12[0];
    consts[36] = We22[0] * ae22[0];
    consts[37] = We22[1] * ae22[0];
  }
}

// ---------------- WMMA GEMM: XL[N,128] = Xh[N,64] @ Wt (f16) ----------------
// One 16x16 output tile per wave; K=64 via two v_wmma_f32_16x16x32_f16.
// All fragment loads are contiguous b128s (Xh row-major, Wt col-major).
// Grid: 3125 blocks x 256 threads (8 waves) == 25000 tiles == (50000/16)*(128/16).
__global__ __launch_bounds__(256)
void k_gemm_xw(const _Float16* __restrict__ Xh, const _Float16* __restrict__ Wt,
               float* __restrict__ XL) {
  const int wave = threadIdx.x >> 5;
  const int lane = threadIdx.x & 31;
  const int tile = blockIdx.x * 8 + wave;
  const int tm = (tile >> 3) << 4;   // row tile * 16
  const int tn = (tile & 7) << 4;    // col tile * 16
  const int hi = lane >> 4;          // lane half
  const int r  = lane & 15;

  v8f acc = {};
  const _Float16* arow = Xh + (size_t)(tm + r) * INF;
  const _Float16* bcol = Wt + (size_t)(tn + r) * INF;
#pragma unroll
  for (int kk = 0; kk < INF; kk += 32) {
    // A fragment (16x32): elem i <-> K = kk + hi*16 + i, row = tm + r
    // -> one contiguous 32B load
    v16h a = *(const v16h*)(arow + kk + hi * 16);
    // B fragment (32x16): elem i <-> K = kk + hi*16 + i, col = tn + r
    // Wt col-major -> one contiguous 32B load
    v16h b = *(const v16h*)(bcol + kk + hi * 16);
    acc = __builtin_amdgcn_wmma_f32_16x16x32_f16(false, a, false, b, (short)0, acc,
                                                 false, false);
  }
  // D layout: VGPR j -> row tm + hi*8 + j, col tn + r
#pragma unroll
  for (int j = 0; j < 8; ++j)
    XL[(size_t)(tm + hi * 8 + j) * HCF + tn + r] = acc[j];
}

// ---------------- per-node attention logits al/ar ----------------
__global__ void k_attn(const float* __restrict__ XL, const float* __restrict__ asrc,
                       const float* __restrict__ adst, float* __restrict__ AL,
                       float* __restrict__ AR, long NN) {
  long id = (long)blockIdx.x * blockDim.x + threadIdx.x;
  if (id >= NN * 8) return;
  long n = id >> 3;
  int h = (int)(id & 7);
  const float* x = XL + n * HCF + h * 16;
  float s = 0.f, t = 0.f;
#pragma unroll
  for (int c = 0; c < 16; ++c) {
    s += x[c] * asrc[h * 16 + c];
    t += x[c] * adst[h * 16 + c];
  }
  AL[id] = s;
  AR[id] = t;
}

// ---------------- layer-1 edge pass 1: logits + segment max ----------------
__global__ void k_edge1(const int* __restrict__ src, const int* __restrict__ dst,
                        long E, long NN, const float* __restrict__ ea,
                        const float* __restrict__ consts, int werOff, float invE,
                        const float* __restrict__ AL, const float* __restrict__ AR,
                        unsigned* __restrict__ M, float* __restrict__ ALPHA) {
  long e = (long)blockIdx.x * blockDim.x + threadIdx.x;
  if (e >= E + NN) return;
  int s, d;
  float ea0 = 0.f, ea1 = 0.f;
  if (e < E) {
    s = src[e]; d = dst[e];
    if (ea) { ea0 = ea[2 * e]; ea1 = ea[2 * e + 1]; }
  } else {
    s = d = (int)(e - E);
    if (ea) { ea0 = consts[0] * invE; ea1 = consts[1] * invE; }   // fill='mean'
  }
#pragma unroll
  for (int h = 0; h < 8; ++h) {
    float a = AL[(size_t)s * 8 + h] + AR[(size_t)d * 8 + h];
    if (ea) a += ea0 * consts[werOff + h] + ea1 * consts[werOff + 8 + h];
    a = lrelu(a);
    ALPHA[e * 8 + h] = a;
    atomicMax(&M[(size_t)d * 8 + h], fkey(a));
  }
}

// ---------------- layer-1 edge pass 2: exp, segment sum, weighted aggregate ----------------
__global__ void k_edge2(const int* __restrict__ src, const int* __restrict__ dst,
                        long E, long NN, const float* __restrict__ XL,
                        const unsigned* __restrict__ M, const float* __restrict__ ALPHA,
                        float* __restrict__ S, float* __restrict__ Xc, int colOff) {
  long id = (long)blockIdx.x * blockDim.x + threadIdx.x;
  if (id >= (E + NN) * 8) return;
  long e = id >> 3;
  int h = (int)(id & 7);
  int s_, d_;
  if (e < E) { s_ = src[e]; d_ = dst[e]; }
  else       { s_ = d_ = (int)(e - E); }
  float ex = __expf(ALPHA[id] - fdec(M[(size_t)d_ * 8 + h]));
  atomicAdd(&S[(size_t)d_ * 8 + h], ex);
  const float* xs = XL + (size_t)s_ * HCF + h * 16;
  __builtin_prefetch(xs, 0, 0);
  float* o = Xc + (size_t)d_ * HIDF + colOff + h * 16;
#pragma unroll
  for (int c = 0; c < 16; ++c) atomicAdd(&o[c], ex * xs[c]);
}

// ---------------- layer-1 normalize + bias + relu ----------------
__global__ void k_norm_relu(const float* __restrict__ S, const float* __restrict__ b,
                            float* __restrict__ Xc, int colOff, long NN) {
  long id = (long)blockIdx.x * blockDim.x + threadIdx.x;
  if (id >= NN * 8) return;
  long n = id >> 3;
  int h = (int)(id & 7);
  float inv = 1.f / (S[id] + 1e-16f);
  float* o = Xc + n * HIDF + colOff + h * 16;
#pragma unroll
  for (int c = 0; c < 16; ++c) {
    float v = o[c] * inv + b[h * 16 + c];
    o[c] = v > 0.f ? v : 0.f;
  }
}

// ---------------- layer-2 GEMV 384->1 (wave per node, coalesced) ----------------
__global__ __launch_bounds__(256)
void k_gemv_l2(const float* __restrict__ Xc, const float* __restrict__ W,
               const float* __restrict__ asrc, const float* __restrict__ adst,
               float* __restrict__ xl2, float* __restrict__ al2,
               float* __restrict__ ar2, long NN) {
  long wid = ((long)blockIdx.x * blockDim.x + threadIdx.x) >> 5;
  int lane = threadIdx.x & 31;
  if (wid >= NN) return;
  const float* row = Xc + wid * HIDF;
  float acc = 0.f;
#pragma unroll
  for (int k = lane; k < HIDF; k += 32) acc += row[k] * W[k];
  for (int o = 16; o > 0; o >>= 1) acc += __shfl_down(acc, o);
  if (lane == 0) {
    xl2[wid] = acc;
    al2[wid] = acc * asrc[0];
    ar2[wid] = acc * adst[0];
  }
}

__global__ void k_edge1_l2(const int* __restrict__ src, const int* __restrict__ dst,
                           long E, long NN, const float* __restrict__ ea,
                           const float* __restrict__ consts, int werOff, float invE,
                           const float* __restrict__ al2, const float* __restrict__ ar2,
                           unsigned* __restrict__ M2, float* __restrict__ A2) {
  long e = (long)blockIdx.x * blockDim.x + threadIdx.x;
  if (e >= E + NN) return;
  int s, d;
  float ea0 = 0.f, ea1 = 0.f;
  if (e < E) {
    s = src[e]; d = dst[e];
    if (ea) { ea0 = ea[2 * e]; ea1 = ea[2 * e + 1]; }
  } else {
    s = d = (int)(e - E);
    if (ea) { ea0 = consts[0] * invE; ea1 = consts[1] * invE; }
  }
  float a = al2[s] + ar2[d];
  if (ea) a += ea0 * consts[werOff] + ea1 * consts[werOff + 1];
  a = lrelu(a);
  A2[e] = a;
  atomicMax(&M2[d], fkey(a));
}

__global__ void k_edge2_l2(const int* __restrict__ src, const int* __restrict__ dst,
                           long E, long NN, const float* __restrict__ xl2,
                           const unsigned* __restrict__ M2, const float* __restrict__ A2,
                           float* __restrict__ S2, float* __restrict__ ACC2) {
  long e = (long)blockIdx.x * blockDim.x + threadIdx.x;
  if (e >= E + NN) return;
  int s, d;
  if (e < E) { s = src[e]; d = dst[e]; }
  else       { s = d = (int)(e - E); }
  float ex = __expf(A2[e] - fdec(M2[d]));
  atomicAdd(&S2[d], ex);
  atomicAdd(&ACC2[d], ex * xl2[s]);
}

__global__ void k_fin_l2(const float* __restrict__ ACC2, const float* __restrict__ S2,
                         const float* __restrict__ b, float* __restrict__ Y, int g,
                         long NN) {
  long n = (long)blockIdx.x * blockDim.x + threadIdx.x;
  if (n >= NN) return;
  Y[n * 3 + g] = ACC2[n] / (S2[n] + 1e-16f) + b[0];
}

__global__ void k_final(const float* __restrict__ Y, const float* __restrict__ lnw,
                        const float* __restrict__ lnb, float* __restrict__ out, long NN) {
  long n = (long)blockIdx.x * blockDim.x + threadIdx.x;
  if (n >= NN) return;
  out[n] = Y[n * 3] * lnw[0] + Y[n * 3 + 1] * lnw[1] + Y[n * 3 + 2] * lnw[2] + lnb[0];
}

// ======================= host launch =======================
static inline int gsz(long n, int b) { return (int)((n + b - 1) / b); }

extern "C" void kernel_launch(void* const* d_in, const int* in_sizes, int n_in,
                              void* d_out, int out_size, void* d_ws, size_t ws_size,
                              hipStream_t stream) {
  const float* X  = (const float*)d_in[0];
  const int*   A[3] = {(const int*)d_in[1], (const int*)d_in[2], (const int*)d_in[3]};
  const float* EA = (const float*)d_in[4];

  const float* W1[3]   = {(const float*)d_in[5],  (const float*)d_in[11], (const float*)d_in[17]};
  const float* AS1[3]  = {(const float*)d_in[6],  (const float*)d_in[12], (const float*)d_in[18]};
  const float* AD1[3]  = {(const float*)d_in[7],  (const float*)d_in[13], (const float*)d_in[19]};
  const float* B1[3]   = {(const float*)d_in[10], (const float*)d_in[16], (const float*)d_in[20]};
  const float* We11 = (const float*)d_in[8],  *Ae11 = (const float*)d_in[9];
  const float* We21 = (const float*)d_in[14], *Ae21 = (const float*)d_in[15];

  const float* W2[3]  = {(const float*)d_in[21], (const float*)d_in[27], (const float*)d_in[33]};
  const float* AS2[3] = {(const float*)d_in[22], (const float*)d_in[28], (const float*)d_in[34]};
  const float* AD2[3] = {(const float*)d_in[23], (const float*)d_in[29], (const float*)d_in[35]};
  const float* B2[3]  = {(const float*)d_in[26], (const float*)d_in[32], (const float*)d_in[36]};
  const float* We12 = (const float*)d_in[24], *Ae12 = (const float*)d_in[25];
  const float* We22 = (const float*)d_in[30], *Ae22 = (const float*)d_in[31];

  const float* LNW = (const float*)d_in[37];
  const float* LNB = (const float*)d_in[38];

  const long NN = (long)in_sizes[0] / INF;       // 50000
  const long E  = (long)in_sizes[1] / 2;         // 800000
  const float invE = 1.0f / (float)E;
  const long EN = E + NN;

  // workspace carve-up
  char* base = (char*)d_ws;
  size_t off = 0;
  auto carve = [&](size_t bytes) { size_t o = off; off = (off + bytes + 255) & ~(size_t)255; return o; };
  float*     XL     = (float*)(base + carve(NN * HCF * 4));
  float*     Xc     = (float*)(base + carve(NN * HIDF * 4));
  _Float16*  Xh     = (_Float16*)(base + carve(NN * INF * 2));
  _Float16*  Wt     = (_Float16*)(base + carve(HCF * INF * 2));
  float*     AL     = (float*)(base + carve(NN * 8 * 4));
  float*     AR     = (float*)(base + carve(NN * 8 * 4));
  unsigned*  M      = (unsigned*)(base + carve(NN * 8 * 4));
  float*     S      = (float*)(base + carve(NN * 8 * 4));
  float*     ALPHA  = (float*)(base + carve(EN * 8 * 4));
  float*     CONSTS = (float*)(base + carve(64 * 4));
  float*     xl2    = (float*)(base + carve(NN * 4));
  float*     al2    = (float*)(base + carve(NN * 4));
  float*     ar2    = (float*)(base + carve(NN * 4));
  unsigned*  M2     = (unsigned*)(base + carve(NN * 4));
  float*     S2     = (float*)(base + carve(NN * 4));
  float*     ACC2   = (float*)(base + carve(NN * 4));
  float*     A2B    = (float*)(base + carve(EN * 4));
  float*     Y      = (float*)(base + carve(NN * 3 * 4));
  (void)ws_size; (void)n_in; (void)out_size;

  const int T = 256;

  // prep: zero consts + Xc, f16 copy of X, edge-feature mean, fold We/a_e
  k_zero<<<1, 64, 0, stream>>>((unsigned*)CONSTS, 64);
  k_zero<<<gsz(NN * HIDF, T), T, 0, stream>>>((unsigned*)Xc, NN * HIDF);
  k_cvt_x<<<gsz(NN * INF, T), T, 0, stream>>>(X, Xh, NN * INF);
  k_reduce_ea<<<256, T, 0, stream>>>(EA, E, CONSTS);
  k_wer<<<1, 64, 0, stream>>>(We11, Ae11, We21, Ae21, We12, Ae12, We22, Ae22, CONSTS);

  const int werOff1[3] = {2, 18, -1};
  const int werOff2[3] = {34, 36, -1};

  // ---- layer 1: three GATConvs (heads=8, C=16, concat into Xc columns) ----
  for (int g = 0; g < 3; ++g) {
    const float* ea = (g < 2) ? EA : nullptr;
    k_cvt_wt<<<gsz(INF * HCF, T), T, 0, stream>>>(W1[g], Wt);
    k_gemm_xw<<<(int)(NN / 16), T, 0, stream>>>(Xh, Wt, XL);
    k_attn<<<gsz(NN * 8, T), T, 0, stream>>>(XL, AS1[g], AD1[g], AL, AR, NN);
    k_zero<<<gsz(NN * 8, T), T, 0, stream>>>(M, NN * 8);
    k_zero<<<gsz(NN * 8, T), T, 0, stream>>>((unsigned*)S, NN * 8);
    k_edge1<<<gsz(EN, T), T, 0, stream>>>(A[g], A[g] + E, E, NN, ea, CONSTS, werOff1[g],
                                          invE, AL, AR, M, ALPHA);
    k_edge2<<<gsz(EN * 8, T), T, 0, stream>>>(A[g], A[g] + E, E, NN, XL, M, ALPHA, S, Xc,
                                              g * HCF);
    k_norm_relu<<<gsz(NN * 8, T), T, 0, stream>>>(S, B1[g], Xc, g * HCF, NN);
  }

  // ---- layer 2: three scalar GATConvs on Xc ----
  for (int g = 0; g < 3; ++g) {
    const float* ea = (g < 2) ? EA : nullptr;
    k_gemv_l2<<<gsz(NN * 32, T), T, 0, stream>>>(Xc, W2[g], AS2[g], AD2[g], xl2, al2,
                                                 ar2, NN);
    k_zero<<<gsz(NN, T), T, 0, stream>>>(M2, NN);
    k_zero<<<gsz(NN, T), T, 0, stream>>>((unsigned*)S2, NN);
    k_zero<<<gsz(NN, T), T, 0, stream>>>((unsigned*)ACC2, NN);
    k_edge1_l2<<<gsz(EN, T), T, 0, stream>>>(A[g], A[g] + E, E, NN, ea, CONSTS,
                                             werOff2[g], invE, al2, ar2, M2, A2B);
    k_edge2_l2<<<gsz(EN, T), T, 0, stream>>>(A[g], A[g] + E, E, NN, xl2, M2, A2B, S2,
                                             ACC2);
    k_fin_l2<<<gsz(NN, T), T, 0, stream>>>(ACC2, S2, B2[g], Y, g, NN);
  }

  k_final<<<gsz(NN, T), T, 0, stream>>>(Y, LNW, LNB, (float*)d_out, NN);
}
